// WindowAttention_5592047420200
// MI455X (gfx1250) — matrix operand order repaired and checked by reading, not compiled, not run on hardware
//
#include <hip/hip_runtime.h>

// ---------------- CDNA5 WMMA types / helpers ----------------
typedef __attribute__((ext_vector_type(16))) _Float16 v16h;
typedef __attribute__((ext_vector_type(8)))  _Float16 v8h;
typedef __attribute__((ext_vector_type(8)))  float    v8f;
typedef __attribute__((ext_vector_type(4)))  unsigned int v4u;
typedef __attribute__((ext_vector_type(8)))  int      v8i;
typedef __attribute__((ext_vector_type(4)))  int      v4i;

__device__ __forceinline__ v8f wmma_f16(v16h a, v16h b, v8f c) {
  // D(f32, 16x16) = A(f16, 16x32) * B(f16, 32x16) + C
  return __builtin_amdgcn_wmma_f32_16x16x32_f16(
      /*neg_a=*/false, a, /*neg_b=*/false, b,
      /*c_mod=*/(short)0, c, /*reuse_a=*/false, /*reuse_b=*/false);
}

// A fragment (16x32, f16): lane 0-15 -> row M=lane, halves 0-15 = K[kb..kb+15]
// with kb = (lane>=16)*16 (contiguous order, matching load_b16).
__device__ __forceinline__ v16h load_a16(const _Float16* base, int ldk) {
  const int lane = threadIdx.x & 31;
  const _Float16* p = base + (lane & 15) * ldk + ((lane >> 4) << 4);
  v8h lo = *(const v8h*)(p);
  v8h hi = *(const v8h*)(p + 8);
  v16h r;
#pragma unroll
  for (int i = 0; i < 8; ++i) { r[i] = lo[i]; r[i + 8] = hi[i]; }
  return r;
}

// B fragment (32x16, f16) from a [N][K] (transposed) buffer:
// lane = column N (mod 16), 16 contiguous K starting at (lane>=16)*16.
__device__ __forceinline__ v16h load_b16(const _Float16* base, int ldk) {
  const int lane = threadIdx.x & 31;
  const _Float16* p = base + (lane & 15) * ldk + ((lane >> 4) << 4);
  v8h lo = *(const v8h*)(p);
  v8h hi = *(const v8h*)(p + 8);
  v16h r;
#pragma unroll
  for (int i = 0; i < 8; ++i) { r[i] = lo[i]; r[i + 8] = hi[i]; }
  return r;
}

// relative position bias index, faithful to the meshgrid('xy') reference
__device__ __forceinline__ int rp_idx(int q, int k) {
  int dy = (q >> 3) - (k >> 3);
  int dx = (q & 7) - (k & 7);
  return (dy + 7) * 15 + (dx + 7);
}

#define NB 4096     // windows (batch)
#define SS 64       // tokens per window
#define CC 256      // channels
#define NH 8        // heads
#define HD 32       // head dim
#define LDA 264     // padded LDS row stride (halfs) for 256-wide tiles
#define PL  72      // padded LDS row stride (halfs) for 64-wide tiles

// ---------------- kernel 0: weights -> transposed f16 ----------------
__global__ __launch_bounds__(256) void wconv_kernel(
    const float* __restrict__ qkv_w, const float* __restrict__ proj_w,
    _Float16* __restrict__ wqkvT, _Float16* __restrict__ wprojT) {
  int i = blockIdx.x * 256 + threadIdx.x;
  if (i < CC * 3 * CC) {                       // qkv_w [256][768] -> [768][256]
    int k = i / (3 * CC), n = i % (3 * CC);
    wqkvT[(size_t)n * CC + k] = (_Float16)qkv_w[i];
  }
  if (i < CC * CC) {                           // proj_w [256][256] -> [256][256]^T
    int k = i / CC, n = i % CC;
    wprojT[(size_t)n * CC + k] = (_Float16)proj_w[i];
  }
}

// ---------------- kernel 1: fused QKV projection ----------------
// one WG (8 waves) per window; out: q (pre-scaled), k, v as [B,H,S,HD] f16
__global__ __launch_bounds__(256) void qkv_kernel(
    const float* __restrict__ x, const float* __restrict__ qkv_b,
    const _Float16* __restrict__ wqkvT,
    _Float16* __restrict__ qf, _Float16* __restrict__ kf, _Float16* __restrict__ vf) {
  __shared__ _Float16 xA[SS * LDA];
  const int b = blockIdx.x, tid = threadIdx.x;
  const float* xb = x + (size_t)b * SS * CC;
  for (int i = tid; i < SS * CC; i += 256)
    xA[(i >> 8) * LDA + (i & 255)] = (_Float16)xb[i];
  __syncthreads();

  const int wave = tid >> 5, lane = tid & 31;
  const int hi8 = (lane >> 4) << 3, nlo = lane & 15;
  const float scale = 0.17677669529663687f;    // 32^-0.5

  for (int ni = 0; ni < 6; ++ni) {
    const int ntile = wave + ni * 8;           // 0..47
    const int n0 = ntile * 16;
    v8f acc[4] = {};
#pragma unroll
    for (int k = 0; k < 8; ++k) {              // K = 256 in steps of 32
      v16h bfr = load_b16(wqkvT + (size_t)n0 * CC + k * 32, CC);
#pragma unroll
      for (int m = 0; m < 4; ++m) {
        v16h afr = load_a16(xA + m * 16 * LDA + k * 32, LDA);
        acc[m] = wmma_f16(afr, bfr, acc[m]);
      }
    }
    // epilogue: +bias, q-scale, scatter to [B,H,S,HD] f16
    const int which = n0 >> 8;                 // 0=q 1=k 2=v
    const int nc = n0 & 255, h = nc >> 5, d0 = nc & 31;
    _Float16* ob = (which == 0) ? qf : (which == 1) ? kf : vf;
    const float sc = (which == 0) ? scale : 1.0f;
    const float bias = qkv_b[n0 + nlo];
#pragma unroll
    for (int m = 0; m < 4; ++m) {
#pragma unroll
      for (int r = 0; r < 8; ++r) {
        const int M = m * 16 + r + hi8;
        ob[(((size_t)b * NH + h) * SS + M) * HD + d0 + nlo] =
            (_Float16)((acc[m][r] + bias) * sc);
      }
    }
  }
}

// ---------------- kernel 2: attention core ----------------
// 4 heads per WG (one wave per head). grid = 2*NB.
__global__ __launch_bounds__(128) void attn_kernel(
    const _Float16* __restrict__ qf, const _Float16* __restrict__ kf,
    const _Float16* __restrict__ vf, const float* __restrict__ mask,
    const float* __restrict__ bias_table, _Float16* __restrict__ aout) {
  __shared__ _Float16 pbuf[4][SS * PL];        // exp(logits), unnormalized
  __shared__ _Float16 vtbuf[4][HD * PL];       // V^T [d][s]
  __shared__ float    rinv[4][SS];             // 1 / rowsum
  const int b = blockIdx.x >> 1;
  const int wave = threadIdx.x >> 5, lane = threadIdx.x & 31;
  const int h = ((blockIdx.x & 1) << 2) + wave;
  const size_t bh = (size_t)b * NH + h;
  const _Float16* qh = qf + bh * (SS * HD);
  const _Float16* kh = kf + bh * (SS * HD);
  const _Float16* vh = vf + bh * (SS * HD);
  const int hi8 = (lane >> 4) << 3, nlo = lane & 15;

  // stage V transposed
  for (int i = lane; i < SS * HD; i += 32) {
    int s = i >> 5, d = i & 31;
    vtbuf[wave][d * PL + s] = vh[i];
  }
  // Q fragments (q already scaled)
  v16h qfr[4];
#pragma unroll
  for (int m = 0; m < 4; ++m) qfr[m] = load_a16(qh + m * 16 * HD, HD);

  // S = Q K^T  (+bias +mask), exp -> pbuf
#pragma unroll
  for (int nt = 0; nt < 4; ++nt) {
    v16h kfr = load_b16(kh + nt * 16 * HD, HD);   // k rows are [s][d] == B layout
    v8f c[4] = {};
#pragma unroll
    for (int m = 0; m < 4; ++m) c[m] = wmma_f16(qfr[m], kfr, c[m]);
#pragma unroll
    for (int m = 0; m < 4; ++m) {
#pragma unroll
      for (int r = 0; r < 8; ++r) {
        const int M = m * 16 + r + hi8;
        const int N = nt * 16 + nlo;
        float logit = c[m][r] + bias_table[rp_idx(M, N) * NH + h] +
                      mask[((size_t)b * SS + M) * SS + N];
        pbuf[wave][M * PL + N] = (_Float16)__expf(logit);
      }
    }
  }
  __syncthreads();
  // row sums (normalization deferred until after PV)
  for (int rr = lane; rr < SS; rr += 32) {
    float s = 0.f;
#pragma unroll 8
    for (int kcol = 0; kcol < SS; ++kcol) s += (float)pbuf[wave][rr * PL + kcol];
    rinv[wave][rr] = 1.0f / s;
  }
  __syncthreads();

  // O = P V   (M=64, N=32, K=64)
  v8f o[4][2] = {};
#pragma unroll
  for (int kt = 0; kt < 2; ++kt) {
    v16h bf0 = load_b16(&vtbuf[wave][0 * PL + kt * 32], PL);
    v16h bf1 = load_b16(&vtbuf[wave][16 * PL + kt * 32], PL);
#pragma unroll
    for (int m = 0; m < 4; ++m) {
      v16h af = load_a16(&pbuf[wave][m * 16 * PL + kt * 32], PL);
      o[m][0] = wmma_f16(af, bf0, o[m][0]);
      o[m][1] = wmma_f16(af, bf1, o[m][1]);
    }
  }
  // normalize rows, write [B,S,C] f16 (head-interleaved)
#pragma unroll
  for (int m = 0; m < 4; ++m) {
#pragma unroll
    for (int r = 0; r < 8; ++r) {
      const int M = m * 16 + r + hi8;
      const float ri = rinv[wave][M];
      const size_t rowbase = ((size_t)b * SS + M) * CC + h * HD;
#pragma unroll
      for (int nt = 0; nt < 2; ++nt)
        aout[rowbase + nt * 16 + nlo] = (_Float16)(o[m][nt][r] * ri);
    }
  }
}

// ---------------- kernel 3: output projection ----------------
// A-panel staged into padded LDS by the Tensor Data Mover (TDM):
// 2D tile 256x64 halfs, pad 4 DWORDs after every 128 DWORDs -> LDA=264 rows.
__global__ __launch_bounds__(256) void proj_kernel(
    const _Float16* __restrict__ aout, const _Float16* __restrict__ wprojT,
    const float* __restrict__ proj_b, float* __restrict__ out) {
  __shared__ _Float16 aA[SS * LDA];
  const int b = blockIdx.x, tid = threadIdx.x;
  const _Float16* ab = aout + (size_t)b * SS * CC;

  if (tid < 32) {  // one wave issues the TDM descriptor (EXEC ignored by TDM)
    unsigned long long ga = (unsigned long long)(uintptr_t)ab;
    unsigned int lds = (unsigned int)(uintptr_t)(&aA[0]);  // low 32b = LDS byte addr
    v4u g0;
    g0[0] = 1u;                                        // count=1, user descriptor
    g0[1] = lds;                                       // lds_addr
    g0[2] = (unsigned int)(ga & 0xFFFFFFFFu);          // global_addr[31:0]
    g0[3] = (unsigned int)((ga >> 32) & 0x01FFFFFFu)   // global_addr[56:32]
            | (2u << 30);                              // type = 2 (image)
    v8i g1;
    g1[0] = (int)((1u << 16)        // data_size = 1 -> 2 bytes
                | (1u << 20)        // pad_enable
                | (6u << 22)        // pad_interval: 128 DWORDs (= one 256-half row)
                | (3u << 25));      // pad_amount:   4 DWORDs (= 8 halfs)
    g1[1] = (int)(256u << 16);      // tensor_dim0 = 256 (lo16 in bits 63:48)
    g1[2] = (int)(64u << 16);       // tensor_dim1 = 64  (lo16 in bits 95:80)
    g1[3] = (int)(256u << 16);      // tile_dim0 = 256   (bits 127:112)
    g1[4] = (int)64;                // tile_dim1 = 64    (bits 143:128)
    g1[5] = (int)256;               // tensor_dim0_stride = 256 (lo32)
    g1[6] = 0;
    g1[7] = 0;
    v4i g2 = {0, 0, 0, 0};          // 2D tensor: dims 2-4 unused
    v4i g3 = {0, 0, 0, 0};
    v8i g4 = {0, 0, 0, 0, 0, 0, 0, 0};  // unused trailing group (6-arg builtin)
    __builtin_amdgcn_tensor_load_to_lds(g0, g1, g2, g3, g4, 0);
    __builtin_amdgcn_s_wait_tensorcnt(0);
  }
  __syncthreads();

  const int wave = tid >> 5, lane = tid & 31;
  const int hi8 = (lane >> 4) << 3, nlo = lane & 15;
  for (int ni = 0; ni < 2; ++ni) {
    const int n0 = (wave * 2 + ni) * 16;
    v8f acc[4] = {};
#pragma unroll
    for (int k = 0; k < 8; ++k) {
      v16h bfr = load_b16(wprojT + (size_t)n0 * CC + k * 32, CC);
#pragma unroll
      for (int m = 0; m < 4; ++m) {
        v16h afr = load_a16(aA + m * 16 * LDA + k * 32, LDA);
        acc[m] = wmma_f16(afr, bfr, acc[m]);
      }
    }
    const float bias = proj_b[n0 + nlo];
#pragma unroll
    for (int m = 0; m < 4; ++m) {
#pragma unroll
      for (int r = 0; r < 8; ++r) {
        const int M = m * 16 + r + hi8;
        out[((size_t)b * SS + M) * CC + n0 + nlo] = acc[m][r] + bias;
      }
    }
  }
}

// ---------------- launch ----------------
extern "C" void kernel_launch(void* const* d_in, const int* in_sizes, int n_in,
                              void* d_out, int out_size, void* d_ws, size_t ws_size,
                              hipStream_t stream) {
  (void)in_sizes; (void)n_in; (void)out_size; (void)ws_size;
  const float* x          = (const float*)d_in[0];
  const float* mask       = (const float*)d_in[1];
  const float* qkv_w      = (const float*)d_in[2];
  const float* qkv_b      = (const float*)d_in[3];
  const float* proj_w     = (const float*)d_in[4];
  const float* proj_b     = (const float*)d_in[5];
  const float* bias_table = (const float*)d_in[6];
  float* out = (float*)d_out;

  const size_t QN = (size_t)NB * NH * SS * HD;  // 67,108,864 halfs per tensor
  _Float16* ws = (_Float16*)d_ws;
  _Float16* qf     = ws;
  _Float16* kf     = ws + QN;
  _Float16* vf     = ws + 2 * QN;
  _Float16* aoutp  = ws + 3 * QN;
  _Float16* wqkvT  = ws + 4 * QN;
  _Float16* wprojT = wqkvT + (size_t)(3 * CC) * CC;

  wconv_kernel<<<(CC * 3 * CC + 255) / 256, 256, 0, stream>>>(qkv_w, proj_w, wqkvT, wprojT);
  qkv_kernel<<<NB, 256, 0, stream>>>(x, qkv_b, wqkvT, qf, kf, vf);
  attn_kernel<<<NB * 2, 128, 0, stream>>>(qf, kf, vf, mask, bias_table, aoutp);
  proj_kernel<<<NB, 256, 0, stream>>>(aoutp, wprojT, proj_b, out);
}
